// MultiHeadAttention_75900662055600
// MI455X (gfx1250) — matrix operand order repaired and checked by reading, not compiled, not run on hardware
//
#include <hip/hip_runtime.h>

// Problem constants (from the reference): B=2, S=2048, D=1024, H=16, DK=64.
constexpr int kB  = 2;
constexpr int kS  = 2048;
constexpr int kD  = 1024;
constexpr int kH  = 16;
constexpr int kDK = 64;

typedef __attribute__((ext_vector_type(16))) _Float16     v16h;
typedef __attribute__((ext_vector_type(8)))  float        v8f;
typedef __attribute__((ext_vector_type(4)))  unsigned int v4u;
typedef __attribute__((ext_vector_type(8)))  int          v8i;
typedef __attribute__((ext_vector_type(4)))  int          v4i;

#ifndef __has_builtin
#define __has_builtin(x) 0
#endif
#if __has_builtin(__builtin_amdgcn_tensor_load_to_lds)
#define HAVE_TDM 1
#endif

static __device__ __forceinline__ v8f wmma16(v16h a, v16h b, v8f c) {
  // D = A(16x32 f16) * B(32x16 f16) + C(16x16 f32)
  return __builtin_amdgcn_wmma_f32_16x16x32_f16(
      /*neg_a=*/false, a, /*neg_b=*/false, b,
      /*c_mod=*/(short)0, c, /*reuse_a=*/false, /*reuse_b=*/false);
}

// K offset of dword-pair p (0..7) in the f16 A-fragment for this lane.
// CDNA5 ISA "16-bit A-Matrix 16x32": lanes 0-15 hold K={0..7,16..23},
// lanes 16-31 hold K={8..15,24..31}; pairs are contiguous (K, K+1).
static __device__ __forceinline__ int a_k(int lane, int p) {
  return ((lane & 16) ? 8 : 0) + ((p < 4) ? (2 * p) : (16 + 2 * (p - 4)));
}

// Build an A-fragment (16x32) from a row-major MxK matrix (f32 or f16 source).
template <typename T>
static __device__ __forceinline__ v16h load_a(const T* src, int ld, int row,
                                              int k0, int lane) {
  v16h a;
  const T* r = src + (size_t)row * ld + k0;
#pragma unroll
  for (int p = 0; p < 8; ++p) {
    int k = a_k(lane, p);
    a[2 * p]     = (_Float16)r[k];
    a[2 * p + 1] = (_Float16)r[k + 1];
  }
  return a;
}

#ifdef HAVE_TDM
// TDM: async-copy one 16-key x 64-dk f16 tile (rows contiguous, stride 64)
// from global memory into LDS. Descriptor per CDNA5 ISA.
static __device__ __forceinline__ void tdm_load_tile(const _Float16* gsrc,
                                                     _Float16* ldst) {
  const unsigned long long ga = (unsigned long long)(uintptr_t)gsrc;
  const unsigned int       la = (unsigned int)(uintptr_t)ldst;
  v4u g0;
  g0[0] = 1u;                                   // count=1, user descriptor
  g0[1] = la;                                   // lds_addr (bytes)
  g0[2] = (unsigned int)(ga & 0xffffffffull);   // global_addr[31:0]
  g0[3] = (unsigned int)((ga >> 32) & 0x01ffffffull)  // global_addr[56:32]
          | (2u << 30);                          // type = 2 ("image")
  v8i g1;
  g1[0] = (int)(1u << 16);      // workgroup_mask=0, data_size=1 (2 bytes)
  g1[1] = (int)(64u << 16);     // tensor_dim0 = 64 (bits 79:48, low half)
  g1[2] = (int)(2048u << 16);   // tensor_dim0 hi=0; tensor_dim1 = 2048 (lo)
  g1[3] = (int)(64u << 16);     // tensor_dim1 hi=0; tile_dim0 = 64
  g1[4] = 16;                   // tile_dim1 = 16 rows, tile_dim2 = 0
  g1[5] = 64;                   // tensor_dim0_stride = 64 elements
  g1[6] = 0;                    // stride hi, tensor_dim1_stride lo
  g1[7] = 0;
  v4i z4 = {0, 0, 0, 0};        // groups 2/3 unused (<=2D tensor)
#if defined(__clang_major__) && __clang_major__ >= 23
  v8i z8 = {0, 0, 0, 0, 0, 0, 0, 0};
  __builtin_amdgcn_tensor_load_to_lds(g0, g1, z4, z4, z8, 0);
#else
  __builtin_amdgcn_tensor_load_to_lds(g0, g1, z4, z4, 0);
#endif
}
#endif  // HAVE_TDM

// ---------------------------------------------------------------------------
// Kernel 0: one-time weight repack  Wt[k][n] = f16(W[n][k]).
// 4 MB read per matrix; projection B-fragments then load f16 directly:
// 16 loads/lane at stride kD, lanes on consecutive n (no cvts per K-step).
// ---------------------------------------------------------------------------
__global__ __launch_bounds__(256) void mha_cvt_w(const float* __restrict__ W,
                                                 _Float16* __restrict__ Wt) {
  const int idx = blockIdx.x * 256 + threadIdx.x;   // kD*kD elements
  const int k = idx >> 10;
  const int n = idx & 1023;
  Wt[(size_t)k * kD + n] = (_Float16)W[(size_t)n * kD + k];
}

// ---------------------------------------------------------------------------
// Kernel 1: input projection  Yh = f16(X @ W^T + b), head-major output.
// One wave computes a 16x64 output strip (1 A-frag reused by 4 B-frags).
// Grid: 512 blocks x 256 threads = 4096 waves = (4096/16)*(1024/64) tiles.
// ---------------------------------------------------------------------------
__global__ __launch_bounds__(256) void mha_proj_qkv(const float* __restrict__ X,
                                                    const _Float16* __restrict__ Wt,
                                                    const float* __restrict__ bias,
                                                    _Float16* __restrict__ Yh) {
  const int lane   = threadIdx.x & 31;
  const int waveId = blockIdx.x * 8 + (threadIdx.x >> 5);
  const int ntiles = kD / 64;                // 16 column groups
  const int mbase  = (waveId / ntiles) * 16;
  const int nbase  = (waveId % ntiles) * 64;
  const int row    = mbase + (lane & 15);

  v8f acc[4] = {};
  for (int kk = 0; kk < kD; kk += 32) {
    if (kk + 64 < kD)
      __builtin_prefetch(&X[(size_t)row * kD + kk + 64], 0, 1);
    v16h a = load_a(X, kD, row, kk, lane);
    const _Float16* wrow = Wt + (size_t)kk * kD + nbase + (lane & 15);  // lane = column n
#pragma unroll
    for (int f = 0; f < 4; ++f) {
      v16h b;                                   // elements = K in A-fragment order
#pragma unroll
      for (int j = 0; j < 16; ++j)
        b[j] = wrow[(size_t)(a_k(lane, j >> 1) + (j & 1)) * kD + 16 * f];
      acc[f] = wmma16(a, b, acc[f]);
    }
  }
#pragma unroll
  for (int f = 0; f < 4; ++f) {
    const int col = nbase + 16 * f + (lane & 15);   // C: lane = column
    const int h = col / kDK, dk = col % kDK;
    const float bv = bias[col];
#pragma unroll
    for (int i = 0; i < 8; ++i) {                   // C: VGPR i = row (+8 hi)
      const int m  = mbase + i + ((lane & 16) ? 8 : 0);
      const int b_ = m / kS, s = m % kS;
      Yh[(((size_t)b_ * kH + h) * kS + s) * kDK + dk] = (_Float16)(acc[f][i] + bv);
    }
  }
}

// ---------------------------------------------------------------------------
// Kernel 2: scores = scale * (q @ k^T), f32 to the attn region of d_out.
// Block = 4 waves; K-tiles double-buffered into LDS via the Tensor Data
// Mover (wave 0 issues tile i+1 while everyone computes tile i); each wave
// keeps its 2 q A-fragments resident for the full 2048-key sweep.
// Grid: (B*H)*32 blocks x 128 threads.
// ---------------------------------------------------------------------------
__global__ __launch_bounds__(128) void mha_scores(const _Float16* __restrict__ qh,
                                                  const _Float16* __restrict__ kh,
                                                  float* __restrict__ attn) {
  __shared__ __align__(16) _Float16 ldsK[2][16 * kDK];  // 2 x 2 KB tiles
  const int lane  = threadIdx.x & 31;
  const int wave  = threadIdx.x >> 5;
  const int bh    = blockIdx.x >> 5;                 // 32 blocks per (b,h)
  const int grp   = blockIdx.x & 31;
  const int mbase = (grp * 4 + wave) * 16;

  const _Float16* qb = qh + (size_t)bh * kS * kDK;
  const _Float16* kb = kh + (size_t)bh * kS * kDK;
  float* ab = attn + (size_t)bh * kS * kS;

  const int  row = mbase + (lane & 15);
  const v16h aq0 = load_a(qb, kDK, row, 0,  lane);
  const v16h aq1 = load_a(qb, kDK, row, 32, lane);
  const float scale = 0.125f;                        // 1/sqrt(DK)
  const int ntile = kS / 16;                         // 128

#ifdef HAVE_TDM
  if (wave == 0) tdm_load_tile(kb, &ldsK[0][0]);     // prime the pipeline
#endif
  for (int i = 0; i < ntile; ++i) {
    const int n0 = i * 16;
#ifdef HAVE_TDM
    if (wave == 0) {
      if (i + 1 < ntile) {
        tdm_load_tile(kb + (size_t)(n0 + 16) * kDK, &ldsK[(i + 1) & 1][0]);
        __builtin_amdgcn_s_wait_tensorcnt(1);        // tile i landed (in-order)
      } else {
        __builtin_amdgcn_s_wait_tensorcnt(0);
      }
    }
    __syncthreads();                                 // tile i visible to all
    const _Float16* kt = &ldsK[i & 1][0];
#else
    __syncthreads();
    ((uint4*)&ldsK[0][0])[threadIdx.x] =
        ((const uint4*)(kb + (size_t)n0 * kDK))[threadIdx.x];
    __syncthreads();
    const _Float16* kt = &ldsK[0][0];
#endif
    v16h b0, b1;                                     // B(d, n) = k[n][d]
#pragma unroll
    for (int j = 0; j < 16; ++j) {
      b0[j] = kt[(lane & 15) * kDK + a_k(lane, j >> 1) + (j & 1)];
      b1[j] = kt[(lane & 15) * kDK + 32 + a_k(lane, j >> 1) + (j & 1)];
    }
    v8f c = {};
    c = wmma16(aq0, b0, c);
    c = wmma16(aq1, b1, c);
#pragma unroll
    for (int r = 0; r < 8; ++r) {
      const int m = mbase + r + ((lane & 16) ? 8 : 0);
      ab[(size_t)m * kS + n0 + (lane & 15)] = c[r] * scale;
    }
#ifdef HAVE_TDM
    __syncthreads();   // everyone done with buf[i&1] before it is re-targeted
#endif
  }
}

// ---------------------------------------------------------------------------
// Kernel 3: row softmax in place. One wave per row; the whole 2048-float row
// lives in 16 float4/lane; shfl_xor reductions over the 32-lane wave.
// Grid: 8192 blocks x 256 threads (8 waves) = 65536 rows.
// ---------------------------------------------------------------------------
__global__ __launch_bounds__(256) void mha_softmax(float* __restrict__ attn) {
  const int lane = threadIdx.x & 31;
  const size_t rowIdx = (size_t)blockIdx.x * 8 + (threadIdx.x >> 5);
  float4* rp = (float4*)(attn + rowIdx * (size_t)kS);

  float4 v[16];
  float mx = -3.402823466e+38f;
#pragma unroll
  for (int i = 0; i < 16; ++i) {
    v[i] = rp[lane + 32 * i];
    mx = fmaxf(mx, fmaxf(fmaxf(v[i].x, v[i].y), fmaxf(v[i].z, v[i].w)));
  }
#pragma unroll
  for (int d = 16; d >= 1; d >>= 1) mx = fmaxf(mx, __shfl_xor(mx, d, 32));

  float sum = 0.f;
#pragma unroll
  for (int i = 0; i < 16; ++i) {
    v[i].x = __expf(v[i].x - mx);
    v[i].y = __expf(v[i].y - mx);
    v[i].z = __expf(v[i].z - mx);
    v[i].w = __expf(v[i].w - mx);
    sum += (v[i].x + v[i].y) + (v[i].z + v[i].w);
  }
#pragma unroll
  for (int d = 16; d >= 1; d >>= 1) sum += __shfl_xor(sum, d, 32);

  const float inv = 1.0f / sum;
#pragma unroll
  for (int i = 0; i < 16; ++i) {
    v[i].x *= inv; v[i].y *= inv; v[i].z *= inv; v[i].w *= inv;
    rp[lane + 32 * i] = v[i];
  }
}

// ---------------------------------------------------------------------------
// Kernel 4: out_head = attn @ v.  One wave = 16 q-rows x full DK=64 (4 C
// frags share each A frag). B-fragment of V is 16 f16 loads/lane at stride DK.
// Output re-concatenated head-interleaved: oh[b][s][h*DK+dk] (f16).
// Grid: 512 blocks x 256 threads = 4096 waves = 32*(2048/16).
// ---------------------------------------------------------------------------
__global__ __launch_bounds__(256) void mha_av(const float* __restrict__ attn,
                                              const _Float16* __restrict__ vh,
                                              _Float16* __restrict__ oh) {
  const int lane   = threadIdx.x & 31;
  const int waveId = blockIdx.x * 8 + (threadIdx.x >> 5);
  const int strips = kS / 16;                     // 128
  const int bh     = waveId / strips;
  const int mbase  = (waveId % strips) * 16;
  const int b_ = bh / kH, h = bh % kH;

  const float*    pa = attn + (size_t)bh * kS * kS;
  const _Float16* vb = vh   + (size_t)bh * kS * kDK;
  const int row = mbase + (lane & 15);

  v8f acc[4] = {};
  for (int kk = 0; kk < kS; kk += 32) {
    v16h a = load_a(pa, kS, row, kk, lane);       // f32 attn -> f16 frag
    const _Float16* vrow = vb + (size_t)kk * kDK + (lane & 15);  // lane = dk column
#pragma unroll
    for (int f = 0; f < 4; ++f) {
      v16h bfrag;                                 // elements = key in A-fragment order
#pragma unroll
      for (int j = 0; j < 16; ++j)
        bfrag[j] = vrow[(size_t)(a_k(lane, j >> 1) + (j & 1)) * kDK + 16 * f];
      acc[f] = wmma16(a, bfrag, acc[f]);
    }
  }
#pragma unroll
  for (int f = 0; f < 4; ++f) {
    const int dk = 16 * f + (lane & 15);
#pragma unroll
    for (int i = 0; i < 8; ++i) {
      const int s = mbase + i + ((lane & 16) ? 8 : 0);
      oh[((size_t)b_ * kS + s) * kD + h * kDK + dk] = (_Float16)acc[f][i];
    }
  }
}

// ---------------------------------------------------------------------------
// Kernel 5: final projection  out = oh @ Wo^T + bo  (f32 to d_out).
// ---------------------------------------------------------------------------
__global__ __launch_bounds__(256) void mha_proj_out(const _Float16* __restrict__ Xh,
                                                    const _Float16* __restrict__ Wt,
                                                    const float* __restrict__ bias,
                                                    float* __restrict__ Y) {
  const int lane   = threadIdx.x & 31;
  const int waveId = blockIdx.x * 8 + (threadIdx.x >> 5);
  const int ntiles = kD / 64;
  const int mbase  = (waveId / ntiles) * 16;
  const int nbase  = (waveId % ntiles) * 64;
  const int row    = mbase + (lane & 15);

  v8f acc[4] = {};
  for (int kk = 0; kk < kD; kk += 32) {
    v16h a = load_a(Xh, kD, row, kk, lane);
    const _Float16* wrow = Wt + (size_t)kk * kD + nbase + (lane & 15);
#pragma unroll
    for (int f = 0; f < 4; ++f) {
      v16h b;
#pragma unroll
      for (int j = 0; j < 16; ++j)
        b[j] = wrow[(size_t)(a_k(lane, j >> 1) + (j & 1)) * kD + 16 * f];
      acc[f] = wmma16(a, b, acc[f]);
    }
  }
#pragma unroll
  for (int f = 0; f < 4; ++f) {
    const int col = nbase + 16 * f + (lane & 15);
    const float bv = bias[col];
#pragma unroll
    for (int i = 0; i < 8; ++i) {
      const int m = mbase + i + ((lane & 16) ? 8 : 0);
      Y[(size_t)m * kD + col] = acc[f][i] + bv;
    }
  }
}

// ---------------------------------------------------------------------------
extern "C" void kernel_launch(void* const* d_in, const int* in_sizes, int n_in,
                              void* d_out, int out_size, void* d_ws, size_t ws_size,
                              hipStream_t stream) {
  const float* Q  = (const float*)d_in[0];
  const float* Kx = (const float*)d_in[1];
  const float* V  = (const float*)d_in[2];
  const float* Wq = (const float*)d_in[3];
  const float* bq = (const float*)d_in[4];
  const float* Wk = (const float*)d_in[5];
  const float* bk = (const float*)d_in[6];
  const float* Wv = (const float*)d_in[7];
  const float* bv = (const float*)d_in[8];
  const float* Wo = (const float*)d_in[9];
  const float* bo = (const float*)d_in[10];

  float* out  = (float*)d_out;                       // (B,S,D)
  float* attn = out + (size_t)kB * kS * kD;          // (B,H,S,S), 537 MB

  // Scratch: q/k/v head-major + head outputs (4 x 8 MB f16) and the four
  // K-major f16 weight repacks (4 x 2 MB) = 40 MB.
  const size_t headElems = (size_t)kB * kH * kS * kDK;
  const size_t wElems    = (size_t)kD * kD;
  _Float16* qh  = (_Float16*)d_ws;
  _Float16* kh  = qh  + headElems;
  _Float16* vh  = kh  + headElems;
  _Float16* oh  = vh  + headElems;
  _Float16* wtq = oh  + headElems;
  _Float16* wtk = wtq + wElems;
  _Float16* wtv = wtk + wElems;
  _Float16* wto = wtv + wElems;

  mha_cvt_w<<<4096, 256, 0, stream>>>(Wq, wtq);
  mha_cvt_w<<<4096, 256, 0, stream>>>(Wk, wtk);
  mha_cvt_w<<<4096, 256, 0, stream>>>(Wv, wtv);
  mha_cvt_w<<<4096, 256, 0, stream>>>(Wo, wto);

  mha_proj_qkv<<<512, 256, 0, stream>>>(Q,  wtq, bq, qh);
  mha_proj_qkv<<<512, 256, 0, stream>>>(Kx, wtk, bk, kh);
  mha_proj_qkv<<<512, 256, 0, stream>>>(V,  wtv, bv, vh);
  mha_scores  <<<1024, 128, 0, stream>>>(qh, kh, attn);
  mha_softmax <<<8192, 256, 0, stream>>>(attn);
  mha_av      <<<512, 256, 0, stream>>>(attn, vh, oh);
  mha_proj_out<<<512, 256, 0, stream>>>(oh, wto, bo, out);

  (void)in_sizes; (void)n_in; (void)out_size; (void)ws_size;
}
